// PureLSTMRegression_12987981103821
// MI455X (gfx1250) — matrix operand order repaired and checked by reading, not compiled, not run on hardware
//
#include <hip/hip_runtime.h>
#include <stdint.h>

#define LAYERS 8
#define H 768
#define B 32
#define T 256
#define G (4 * H)          // 3072 gate rows
#define K2 (2 * H)         // 1536 fused K (x | h)
#define NWGR 24            // recurrent workgroups: 32 hidden units each
#define CW 32              // hidden units per workgroup
#define NSTEPS (K2 / 32)   // 48 WMMA k-steps per timestep (24 x-half + 24 h-half)
#define XROWP (H + 8)      // padded LDS row (bf16 elems): 1552B stride -> conflict-free
#define XBUF (B * XROWP)   // one staged 32xH bf16 matrix (elems)

typedef __attribute__((ext_vector_type(16))) __bf16 v16bf;
typedef __attribute__((ext_vector_type(8)))  float  v8f;

struct Frag32 { uint4 lo, hi; };   // 32 bytes = 16 bf16 (one WMMA operand per lane)

// per-lane async 16B copy: global -> LDS, tracked by ASYNCcnt
#define ASYNC_LD16(ldsptr, gptr)                                            \
  asm volatile("global_load_async_to_lds_b128 %0, %1, off"                  \
               :: "v"((unsigned)(size_t)(ldsptr)), "v"(gptr) : "memory")

static __device__ __forceinline__ unsigned short f32_to_bf16(float f) {
  union { float f; unsigned u; } v; v.f = f;
  unsigned r = 0x7FFFu + ((v.u >> 16) & 1u);   // round-to-nearest-even
  v.u += r;
  return (unsigned short)(v.u >> 16);
}

// ---------------- small prep kernels (re-run every call: determinism) ----------------

__global__ void init_ctrl_kernel(unsigned* bars) {
  if (threadIdx.x < LAYERS) bars[threadIdx.x] = 0u;
}

__global__ void cvt_f32_bf16_kernel(const float* __restrict__ in,
                                    unsigned short* __restrict__ out, int n) {
  int i = blockIdx.x * blockDim.x + threadIdx.x;
  if (i < n) out[i] = f32_to_bf16(in[i]);
}

// Pack [W_ih | W_hh] (layer l) into WMMA-B fragment order:
// elem ((ct*48+ks)*32 + lane)*16 + j  <=  W[col = ct*16 + lane%16][k = 32*ks + 16*(lane/16) + j]
__global__ void pack_w_kernel(const float* __restrict__ Wih, const float* __restrict__ Whh,
                              unsigned short* __restrict__ Wp, int layer) {
  int idx = blockIdx.x * blockDim.x + threadIdx.x;
  if (idx >= G * K2) return;
  int j    = idx & 15;
  int lane = (idx >> 4) & 31;
  int blk  = idx >> 9;               // ct*48 + ks
  int ks   = blk % NSTEPS;
  int ct   = blk / NSTEPS;
  int col  = ct * 16 + (lane & 15);
  int k    = 32 * ks + ((lane >> 4) << 4) + j;
  float w;
  if (k < H) w = Wih[((size_t)layer * G + col) * H + k];
  else       w = Whh[((size_t)layer * G + col) * H + (k - H)];
  Wp[idx] = f32_to_bf16(w);
}

__global__ void bias_kernel(const float* __restrict__ bi, const float* __restrict__ bh,
                            float* __restrict__ bias, int layer) {
  int i = blockIdx.x * blockDim.x + threadIdx.x;
  if (i < G) bias[i] = bi[(size_t)layer * G + i] + bh[(size_t)layer * G + i];
}

// ---------------- persistent per-layer LSTM kernel ----------------

static __device__ __forceinline__ void grid_sync(unsigned* bar, unsigned target) {
  __threadfence();          // release this block's global writes
  __syncthreads();
  if (threadIdx.x == 0) {
    __hip_atomic_fetch_add(bar, 1u, __ATOMIC_ACQ_REL, __HIP_MEMORY_SCOPE_AGENT);
    while (__hip_atomic_load(bar, __ATOMIC_ACQUIRE, __HIP_MEMORY_SCOPE_AGENT) < target) {
      __builtin_amdgcn_s_sleep(2);
    }
  }
  __syncthreads();
  __threadfence();          // acquire for all threads
}

__global__ __launch_bounds__(256) void lstm_layer_kernel(
    const unsigned short* __restrict__ seq_in,   // (B,T,H) bf16 layer input
    unsigned short* __restrict__ seq_out,        // (B,T,H) bf16 layer output
    const unsigned short* __restrict__ Wp,       // packed fused weights (bf16)
    const float* __restrict__ bias,              // b_ih + b_hh (G)
    unsigned short* __restrict__ h_bf,           // (B,H) bf16 broadcast h
    float* __restrict__ h_last,                  // (B,H) fp32 final-step h
    unsigned* __restrict__ bar)                  // grid barrier counter (this layer)
{
  extern __shared__ char smem[];
  unsigned short* lds_X  = (unsigned short*)smem;                 // 2 x (B x XROWP) bf16
  unsigned short* lds_Hs = lds_X + 2 * XBUF;                      // B x XROWP bf16
  float* lds_G = (float*)(smem + (size_t)3 * XBUF * 2);           // B x 4 x CW f32 (16KB)
  float* lds_C = lds_G + B * 4 * CW;                              // B x CW f32 (4KB)

  const int tid  = threadIdx.x;
  const int wg   = blockIdx.x;        // hidden-unit tile (CW units) 0..23
  const int wave = tid >> 5;          // 0..7
  const int lane = tid & 31;
  const int g    = wave & 3;          // gate (i,f,g,o)
  const int nt   = wave >> 2;         // 16-col subtile within the CW slice
  const int ct   = g * (H / 16) + wg * 2 + nt;   // gate-column tile 0..191

  // zero cell state and our slice of the h broadcast buffer
  for (int i = tid; i < B * CW; i += 256) {
    lds_C[i] = 0.f;
    int m = i >> 5, n = i & 31;
    h_bf[(size_t)m * H + wg * CW + n] = 0;
  }
  // preload x_0 into buffer 0 (async; completion enforced inside the loop)
#pragma unroll
  for (int it = 0; it < 12; ++it) {
    int idx = tid + it * 256;
    int m = idx / 96, kc = (idx % 96) * 8;
    ASYNC_LD16(lds_X + (size_t)m * XROWP + kc,
               seq_in + (size_t)m * T * H + kc);
  }
  unsigned sync_no = 1;
  grid_sync(bar, sync_no * NWGR); ++sync_no;

  const float biasv = bias[g * H + wg * CW + nt * 16 + (lane & 15)];
  const int   r0    = lane & 15;                 // tile row for this lane
  const int   csh   = (lane < 16) ? 0 : 16;      // A-fragment K sub-offset

  for (int t = 0; t < T; ++t) {
    const unsigned short* xb = lds_X + (size_t)(t & 1) * XBUF;
    unsigned short*       xn = lds_X + (size_t)((t + 1) & 1) * XBUF;

    // ---- async-stage h_t, then prefetch x_{t+1} behind it ----
#pragma unroll
    for (int it = 0; it < 12; ++it) {
      int idx = tid + it * 256;
      int m = idx / 96, kc = (idx % 96) * 8;
      ASYNC_LD16(lds_Hs + (size_t)m * XROWP + kc,
                 h_bf + (size_t)m * H + kc);
    }
    const int tn = (t + 1) & (T - 1);            // wraps at T-1: harmless re-read of x_0
#pragma unroll
    for (int it = 0; it < 12; ++it) {
      int idx = tid + it * 256;
      int m = idx / 96, kc = (idx % 96) * 8;
      ASYNC_LD16(xn + (size_t)m * XROWP + kc,
                 seq_in + (size_t)m * T * H + (size_t)tn * H + kc);
    }
    // in-order completion: waiting to <=12 retires x_t prefetch + h stage,
    // leaves the 12 x_{t+1} chunks in flight under the K-loop.
    asm volatile("s_wait_asynccnt 12" ::: "memory");
    __syncthreads();

    // ---- fused gate GEMM: two 16x16 M-tiles share each B fragment ----
    v8f acc0, acc1;
#pragma unroll
    for (int r = 0; r < 8; ++r) { acc0[r] = biasv; acc1[r] = biasv; }

    const unsigned short* bptr = Wp + (size_t)ct * NSTEPS * 512 + lane * 16;
#pragma unroll 4
    for (int ks = 0; ks < NSTEPS / 2; ++ks) {    // x half of K
      Frag32 a0, a1, bfr;
      const unsigned short* ap0 = xb + (size_t)r0 * XROWP + ks * 32 + csh;
      const unsigned short* ap1 = ap0 + (size_t)16 * XROWP;
      a0.lo = *(const uint4*)(ap0);  a0.hi = *(const uint4*)(ap0 + 8);
      a1.lo = *(const uint4*)(ap1);  a1.hi = *(const uint4*)(ap1 + 8);
      bfr.lo = *(const uint4*)(bptr); bfr.hi = *(const uint4*)(bptr + 8);
      __builtin_prefetch(bptr + 512, 0, 1);
      acc0 = __builtin_amdgcn_wmma_f32_16x16x32_bf16(
          false, __builtin_bit_cast(v16bf, a0), false, __builtin_bit_cast(v16bf, bfr),
          (short)0, acc0, false, false);
      acc1 = __builtin_amdgcn_wmma_f32_16x16x32_bf16(
          false, __builtin_bit_cast(v16bf, a1), false, __builtin_bit_cast(v16bf, bfr),
          (short)0, acc1, false, false);
      bptr += 512;
    }
#pragma unroll 4
    for (int ks = 0; ks < NSTEPS / 2; ++ks) {    // h half of K
      Frag32 a0, a1, bfr;
      const unsigned short* ap0 = lds_Hs + (size_t)r0 * XROWP + ks * 32 + csh;
      const unsigned short* ap1 = ap0 + (size_t)16 * XROWP;
      a0.lo = *(const uint4*)(ap0);  a0.hi = *(const uint4*)(ap0 + 8);
      a1.lo = *(const uint4*)(ap1);  a1.hi = *(const uint4*)(ap1 + 8);
      bfr.lo = *(const uint4*)(bptr); bfr.hi = *(const uint4*)(bptr + 8);
      __builtin_prefetch(bptr + 512, 0, 1);
      acc0 = __builtin_amdgcn_wmma_f32_16x16x32_bf16(
          false, __builtin_bit_cast(v16bf, a0), false, __builtin_bit_cast(v16bf, bfr),
          (short)0, acc0, false, false);
      acc1 = __builtin_amdgcn_wmma_f32_16x16x32_bf16(
          false, __builtin_bit_cast(v16bf, a1), false, __builtin_bit_cast(v16bf, bfr),
          (short)0, acc1, false, false);
      bptr += 512;
    }

    // ---- exchange gate pre-activations through LDS ----
#pragma unroll
    for (int r = 0; r < 8; ++r) {
      int mm0 = (lane < 16) ? r : r + 8;         // acc0 rows 0..15
      int nn  = nt * 16 + (lane & 15);
      lds_G[(mm0 * 4 + g) * CW + nn]        = acc0[r];
      lds_G[((mm0 + 16) * 4 + g) * CW + nn] = acc1[r];
    }
    __syncthreads();

    // ---- LSTM cell update (B*CW = 1024 cells over 256 threads) ----
    for (int i = tid; i < B * CW; i += 256) {
      int mm = i >> 5, nn = i & 31;
      const float* gp = &lds_G[mm * 4 * CW + nn];
      float gi = gp[0], gf = gp[CW], gg = gp[2 * CW], go = gp[3 * CW];
      float si = 1.f / (1.f + __expf(-gi));
      float sf = 1.f / (1.f + __expf(-gf));
      float so = 1.f / (1.f + __expf(-go));
      float c  = sf * lds_C[i] + si * tanhf(gg);
      lds_C[i] = c;
      float h  = so * tanhf(c);
      int col  = wg * CW + nn;
      unsigned short hb = f32_to_bf16(h);
      h_bf[(size_t)mm * H + col] = hb;                              // recurrence broadcast
      seq_out[(size_t)mm * T * H + (size_t)t * H + col] = hb;       // next layer's input
      if (t == T - 1) h_last[(size_t)mm * H + col] = h;             // fp32 for final GEMM
    }
    __syncthreads();

    // ---- all workgroups agree h is published before next timestep ----
    grid_sync(bar, sync_no * NWGR); ++sync_no;
  }
}

// ---------------- final projection: out = h_last @ W_out^T ----------------

__global__ void out_gemm_kernel(const float* __restrict__ h_last,
                                const float* __restrict__ Wout,
                                float* __restrict__ out) {
  int idx = blockIdx.x * blockDim.x + threadIdx.x;
  if (idx >= B * H) return;
  int b = idx / H, j = idx % H;
  const float4* hv = (const float4*)(h_last + (size_t)b * H);
  const float4* wv = (const float4*)(Wout + (size_t)j * H);
  float s = 0.f;
#pragma unroll 4
  for (int k = 0; k < H / 4; ++k) {
    float4 a = hv[k], w = wv[k];
    s += a.x * w.x + a.y * w.y + a.z * w.z + a.w * w.w;
  }
  out[idx] = s;
}

// ---------------- host orchestration ----------------

extern "C" void kernel_launch(void* const* d_in, const int* in_sizes, int n_in,
                              void* d_out, int out_size, void* d_ws, size_t ws_size,
                              hipStream_t stream) {
  const float* x    = (const float*)d_in[0];
  const float* Wih  = (const float*)d_in[1];
  const float* Whh  = (const float*)d_in[2];
  const float* bi   = (const float*)d_in[3];
  const float* bh   = (const float*)d_in[4];
  const float* Wout = (const float*)d_in[5];
  float* out = (float*)d_out;

  char* ws = (char*)d_ws;
  size_t off = 0;
  auto carve = [&](size_t bytes) { char* p = ws + off; off = (off + bytes + 255) & ~(size_t)255; return p; };
  unsigned short* seqA = (unsigned short*)carve((size_t)B * T * H * 2);
  unsigned short* seqB = (unsigned short*)carve((size_t)B * T * H * 2);
  unsigned short* Wp   = (unsigned short*)carve((size_t)G * K2 * 2);
  float* bias   = (float*)carve((size_t)G * 4);
  float* h_last = (float*)carve((size_t)B * H * 4);
  unsigned short* h_bf = (unsigned short*)carve((size_t)B * H * 2);
  unsigned* bars = (unsigned*)carve(LAYERS * sizeof(unsigned));

  const size_t smem_bytes = (size_t)3 * XBUF * 2       // x double buffer + h stage
                          + (size_t)B * 4 * CW * 4     // gate exchange
                          + (size_t)B * CW * 4;        // cell state

  hipLaunchKernelGGL(init_ctrl_kernel, dim3(1), dim3(64), 0, stream, bars);
  int nx = B * T * H;
  hipLaunchKernelGGL(cvt_f32_bf16_kernel, dim3((nx + 255) / 256), dim3(256), 0, stream, x, seqA, nx);

  for (int l = 0; l < LAYERS; ++l) {
    hipLaunchKernelGGL(pack_w_kernel, dim3((G * K2 + 255) / 256), dim3(256), 0, stream, Wih, Whh, Wp, l);
    hipLaunchKernelGGL(bias_kernel, dim3((G + 255) / 256), dim3(256), 0, stream, bi, bh, bias, l);
    const unsigned short* sin = (l & 1) ? seqB : seqA;
    unsigned short* sout      = (l & 1) ? seqA : seqB;
    hipLaunchKernelGGL(lstm_layer_kernel, dim3(NWGR), dim3(256), smem_bytes, stream,
                       sin, sout, Wp, bias, h_bf, h_last, &bars[l]);
  }
  hipLaunchKernelGGL(out_gemm_kernel, dim3((B * H + 255) / 256), dim3(256), 0, stream,
                     h_last, Wout, out);
}
